// GCN_6158983102812
// MI455X (gfx1250) — matrix operand order repaired and checked by reading, not compiled, not run on hardware
//
#include <hip/hip_runtime.h>

#define IN_FT 256
#define OUT_FT 64

typedef __attribute__((ext_vector_type(16))) __bf16 v16bf;
typedef __attribute__((ext_vector_type(8)))  float  v8f;
typedef __attribute__((ext_vector_type(4)))  float  v4f;

// ---------------------------------------------------------------------------
// Kernel 0: zero the accumulator (d_out is poisoned by the harness)
// ---------------------------------------------------------------------------
__global__ void gcn_zero(float* __restrict__ out, int total) {
    int i = blockIdx.x * blockDim.x + threadIdx.x;
    if (i < total) out[i] = 0.0f;
}

// ---------------------------------------------------------------------------
// Kernel 1: fts[N,64] = seq[N,256] @ W[64,256]^T via v_wmma_f32_16x16x32_bf16
// One wave computes a 16(M) x 64(N) strip; 8 waves per block -> 128 rows/block.
// ---------------------------------------------------------------------------
__global__ __launch_bounds__(256) void gcn_gemm_wmma(
        const float* __restrict__ seq, const float* __restrict__ W,
        float* __restrict__ fts, int n) {
    const int lane = threadIdx.x & 31;
    const int wave = threadIdx.x >> 5;
    const int rsel = lane & 15;          // row/col selector within tile
    const int hi   = lane >> 4;          // 0: lanes 0-15, 1: lanes 16-31

    const int baseRow = (blockIdx.x * 8 + wave) * 16;

    // A-fragment source row for this lane (clamped: keeps loads in-bounds,
    // keeps control flow uniform so EXEC is all-1s at every WMMA).
    int arow = baseRow + rsel;
    arow = arow < n ? arow : (n - 1);
    const float* __restrict__ aptr = seq + (long)arow * IN_FT;

    v8f acc[4];
    acc[0] = (v8f){}; acc[1] = (v8f){}; acc[2] = (v8f){}; acc[3] = (v8f){};

    for (int k0 = 0; k0 < IN_FT; k0 += 32) {
        // ---- A fragment (16x32 bf16), per ISA layout:
        // lanes 0-15 : K = k0+[0..7]  and k0+[16..23]
        // lanes 16-31: K = k0+[8..15] and k0+[24..31]
        const int ka = k0 + hi * 8;
        v4f a0 = *(const v4f*)(aptr + ka);
        v4f a1 = *(const v4f*)(aptr + ka + 4);
        v4f a2 = *(const v4f*)(aptr + ka + 16);
        v4f a3 = *(const v4f*)(aptr + ka + 20);
        v16bf a;
#pragma unroll
        for (int i = 0; i < 4; ++i) {
            a[i]      = (__bf16)a0[i];
            a[4 + i]  = (__bf16)a1[i];
            a[8 + i]  = (__bf16)a2[i];
            a[12 + i] = (__bf16)a3[i];
        }
        if (k0 + 32 < IN_FT)
            __builtin_prefetch(aptr + k0 + 32, 0, 0);   // global_prefetch_b8

        // ---- B fragments: B[k][col] = W[col][k], col = nt*16 + rsel
        // lanes 0-15 hold K = k0+[0..7] and k0+[16..23], lanes 16-31 hold K = k0+[8..15] and k0+[24..31]
        const int kb = k0 + hi * 8;
#pragma unroll
        for (int nt = 0; nt < 4; ++nt) {
            const float* __restrict__ wp = W + (long)(nt * 16 + rsel) * IN_FT + kb;
            v4f b0 = *(const v4f*)(wp);
            v4f b1 = *(const v4f*)(wp + 4);
            v4f b2 = *(const v4f*)(wp + 16);
            v4f b3 = *(const v4f*)(wp + 20);
            v16bf b;
#pragma unroll
            for (int i = 0; i < 4; ++i) {
                b[i]      = (__bf16)b0[i];
                b[4 + i]  = (__bf16)b1[i];
                b[8 + i]  = (__bf16)b2[i];
                b[12 + i] = (__bf16)b3[i];
            }
            acc[nt] = __builtin_amdgcn_wmma_f32_16x16x32_bf16(
                /*neg_a=*/false, a, /*neg_b=*/false, b,
                /*c_mod=*/(short)0, acc[nt],
                /*reuse_a=*/false, /*reuse_b=*/false);
        }
    }

    // ---- Store D: VGPR j holds rows (baseRow + j + hi*8), col = nt*16+rsel
#pragma unroll
    for (int nt = 0; nt < 4; ++nt) {
#pragma unroll
        for (int j = 0; j < 8; ++j) {
            int row = baseRow + hi * 8 + j;
            if (row < n)
                fts[(long)row * OUT_FT + nt * 16 + rsel] = acc[nt][j];
        }
    }
}

// ---------------------------------------------------------------------------
// Kernel 2: edge scatter. One 64-lane group per edge, one feature per lane.
// fts (12.8 MB) is L2-resident on MI455X (192 MB L2); atomics execute at L2.
// ---------------------------------------------------------------------------
__global__ __launch_bounds__(256) void gcn_edge_scatter(
        const float* __restrict__ fts,
        const int* __restrict__ esrc, const int* __restrict__ edst,
        const float* __restrict__ eval,
        float* __restrict__ out, int e_total) {
    int e = blockIdx.x * 4 + (threadIdx.x >> 6);
    int f = threadIdx.x & 63;
    if (e >= e_total) return;
    int s = esrc[e];
    int d = edst[e];
    float m = fts[(long)s * OUT_FT + f] * eval[e];
    // hardware global_atomic_add_f32 (no return)
    unsafeAtomicAdd(out + (long)d * OUT_FT + f, m);
}

// ---------------------------------------------------------------------------
// Kernel 3: bias + PReLU, in place
// ---------------------------------------------------------------------------
__global__ void gcn_finalize(float* __restrict__ out,
                             const float* __restrict__ bias,
                             const float* __restrict__ alpha, int total) {
    int i = blockIdx.x * blockDim.x + threadIdx.x;
    if (i >= total) return;
    float v = out[i] + bias[i & (OUT_FT - 1)];
    float a = alpha[0];
    out[i] = (v >= 0.0f) ? v : a * v;
}

// ---------------------------------------------------------------------------
extern "C" void kernel_launch(void* const* d_in, const int* in_sizes, int n_in,
                              void* d_out, int out_size, void* d_ws, size_t ws_size,
                              hipStream_t stream) {
    const float* seq   = (const float*)d_in[0];
    const float* W     = (const float*)d_in[1];
    const float* bias  = (const float*)d_in[2];
    const float* alpha = (const float*)d_in[3];
    const int*   esrc  = (const int*)d_in[4];
    const int*   edst  = (const int*)d_in[5];
    const float* eval  = (const float*)d_in[6];

    float* out = (float*)d_out;
    float* fts = (float*)d_ws;           // N*64 floats of scratch

    const int n = in_sizes[0] / IN_FT;   // 50000
    const int e = in_sizes[4];           // 800000
    const int total = n * OUT_FT;

    gcn_zero<<<(total + 255) / 256, 256, 0, stream>>>(out, total);

    const int rowTiles = (n + 15) / 16;
    gcn_gemm_wmma<<<(rowTiles + 7) / 8, 256, 0, stream>>>(seq, W, fts, n);

    gcn_edge_scatter<<<(e + 3) / 4, 256, 0, stream>>>(fts, esrc, edst, eval, out, e);

    gcn_finalize<<<(total + 255) / 256, 256, 0, stream>>>(out, bias, alpha, total);
}
